// TransformerBlock_42082089566324
// MI455X (gfx1250) — matrix operand order repaired and checked by reading, not compiled, not run on hardware
//
#include <hip/hip_runtime.h>
#include <hip/hip_bf16.h>

// ---------------------------------------------------------------------------
// CDNA5 / gfx1250 transformer block forward.
// Matmuls: v_wmma_f32_16x16x32_f16 (wave32 WMMA, f32 accum).
// GEMM staging: double-buffered GLOBAL_LOAD_ASYNC_TO_LDS_B128 (ASYNCcnt).
// ---------------------------------------------------------------------------

typedef __attribute__((ext_vector_type(16))) _Float16     v16h;
typedef __attribute__((ext_vector_type(8)))  float        v8f;
typedef __attribute__((ext_vector_type(8)))  unsigned int v8u;

// Exact pointee type of the async builtin's pointer params (from the clang
// diagnostic: "__attribute__((__vector_size__(4 * sizeof(int)))) int __device__ *")
typedef int v4i_gcc __attribute__((vector_size(16)));
typedef __attribute__((address_space(1))) v4i_gcc* as1_v4i_ptr;  // global
typedef __attribute__((address_space(3))) v4i_gcc* as3_v4i_ptr;  // LDS

#define WMMA_F16(A, B, C) \
  __builtin_amdgcn_wmma_f32_16x16x32_f16(false, (A), false, (B), (short)0, (C), false, false)

// Problem constants
#define TOK   4096   // B*S
#define DIM   1024
#define NHEAD 16
#define HSIZ  64
#define FFDIM 4096
#define SEQ   2048
#define C3    3072   // 3*DIM, packed qkv row stride

// ---------------------------------------------------------------------------
// CDNA5 async copy global -> LDS (16B per lane), guarded for toolchain drift.
// ---------------------------------------------------------------------------
#if defined(__has_builtin)
#if __has_builtin(__builtin_amdgcn_global_load_async_to_lds_b128)
#define HAS_ASYNC_LDS 1
#endif
#endif
#ifndef HAS_ASYNC_LDS
#define HAS_ASYNC_LDS 0
#endif

__device__ inline void async_cp16(const _Float16* g, _Float16* l) {
#if HAS_ASYNC_LDS
  __builtin_amdgcn_global_load_async_to_lds_b128(
      (as1_v4i_ptr)(void*)g, (as3_v4i_ptr)(void*)l, 0, 0);
#else
  *(float4*)l = *(const float4*)g;   // sync fallback through VGPRs
#endif
}

__device__ inline void wait_async0() {
#if HAS_ASYNC_LDS
#if __has_builtin(__builtin_amdgcn_s_wait_asynccnt)
  __builtin_amdgcn_s_wait_asynccnt(0);
#else
  asm volatile("s_wait_asynccnt 0" ::: "memory");
#endif
#endif
}

// ---------------------------------------------------------------------------
// Cross-lane row reductions across the 16 lanes of each wave32 half
// (a WMMA C-layout matrix row lives on exactly one 16-lane half).
// ---------------------------------------------------------------------------
__device__ inline float rowmax16(float v) {
#pragma unroll
  for (int m = 1; m < 16; m <<= 1) v = fmaxf(v, __shfl_xor(v, m, 32));
  return v;
}
__device__ inline float rowsum16(float v) {
#pragma unroll
  for (int m = 1; m < 16; m <<= 1) v += __shfl_xor(v, m, 32);
  return v;
}

// ---------------------------------------------------------------------------
// Weight conversion: fp32 -> f16, stored K-contiguous (transposed) so GEMM B
// tiles are contiguous 16B chunks for the async-to-LDS path.
// ---------------------------------------------------------------------------

// out[n*K + k] = in[k*N + n]   (in: [K,N] row-major fp32)
__global__ void cvt_f32_f16_T(const float* __restrict__ in, _Float16* __restrict__ out,
                              int K, int N) {
  int i = blockIdx.x * 256 + threadIdx.x;
  if (i >= K * N) return;
  int n = i / K, k = i % K;                 // write-coalesced
  out[i] = (_Float16)in[(size_t)k * N + n];
}

// Pack Wq/Wk/Wv [H,D,HS] fp32 into K-contiguous f16 [3*D][D]:
// row c: c/1024 selects q/k/v, (c%1024)/64 = head, c%64 = hs index; col = d.
__global__ void pack_qkv_wT(const float* __restrict__ Wq, const float* __restrict__ Wk,
                            const float* __restrict__ Wv, _Float16* __restrict__ out) {
  int i = blockIdx.x * 256 + threadIdx.x;
  if (i >= DIM * C3) return;
  int c = i / DIM, d = i % DIM;             // write-coalesced
  int which = c >> 10;
  int cc = c & 1023;
  int hh = cc >> 6, e = cc & 63;
  const float* W = (which == 0) ? Wq : ((which == 1) ? Wk : Wv);
  out[i] = (_Float16)W[((size_t)hh * DIM + d) * HSIZ + e];
}

// ---------------------------------------------------------------------------
// LayerNorm: one row (D=1024) per 256-thread block, emits f16 activations.
// ---------------------------------------------------------------------------
__global__ __launch_bounds__(256)
void ln_f16_kernel(const float* __restrict__ x, const float* __restrict__ sc,
                   const float* __restrict__ bi, _Float16* __restrict__ out) {
  int row = blockIdx.x;
  const float* xr = x + (size_t)row * DIM;
  float4 v = ((const float4*)xr)[threadIdx.x];           // 256 * 4 = 1024
  float s  = v.x + v.y + v.z + v.w;
  float ss = v.x * v.x + v.y * v.y + v.z * v.z + v.w * v.w;
#pragma unroll
  for (int m = 1; m < 32; m <<= 1) {                     // wave32 reduce
    s  += __shfl_xor(s, m, 32);
    ss += __shfl_xor(ss, m, 32);
  }
  __shared__ float red[2][8];
  int wid = threadIdx.x >> 5, lane = threadIdx.x & 31;
  if (lane == 0) { red[0][wid] = s; red[1][wid] = ss; }
  __syncthreads();
  if (threadIdx.x == 0) {
    float a = 0.f, b2 = 0.f;
#pragma unroll
    for (int i = 0; i < 8; ++i) { a += red[0][i]; b2 += red[1][i]; }
    red[0][0] = a; red[1][0] = b2;
  }
  __syncthreads();
  float mu   = red[0][0] * (1.0f / DIM);
  float var  = red[1][0] * (1.0f / DIM) - mu * mu;
  float rstd = rsqrtf(var + 1e-5f);
  int c0 = threadIdx.x * 4;
  _Float16* o = out + (size_t)row * DIM + c0;
  float sx[4] = {v.x, v.y, v.z, v.w};
#pragma unroll
  for (int i = 0; i < 4; ++i)
    o[i] = (_Float16)((sx[i] - mu) * rstd * sc[c0 + i] + bi[c0 + i]);
}

// ---------------------------------------------------------------------------
// WMMA GEMM: C[M,N] = A[M,K] * BT[N,K]^T, f16 operands, f32 accumulate.
// Block tile 128x128x32, 256 threads = 8 waves (4x2), wave tile 32x64.
// Double-buffered async-to-LDS staging: tile i+1 streams while tile i runs
// on the matrix pipe; drained with s_wait_asynccnt 0 + barrier.
// Epilogues: 0 = f16 store; 1 = relu(acc+bias) -> f16; 2 = acc+bias+resid -> f32
// ---------------------------------------------------------------------------
enum { EPI_F16 = 0, EPI_BIAS_RELU_F16 = 1, EPI_BIAS_RES_F32 = 2 };

template <int EPI>
__global__ __launch_bounds__(256)
void gemm_wmma(const _Float16* __restrict__ A, const _Float16* __restrict__ BT,
               void* __restrict__ Cout, const float* __restrict__ bias,
               const float* __restrict__ resid, int M, int N, int K) {
  constexpr int BM = 128, BN = 128, BK = 32, LDP = BK + 8;  // halves stride (pad)
  __shared__ __align__(16) _Float16 Asb[2][BM * LDP];       // [row][k]
  __shared__ __align__(16) _Float16 Bsb[2][BN * LDP];       // [n][k] (K-contig)

  const int tid  = threadIdx.x;
  const int wid  = tid >> 5, lane = tid & 31;
  const int ln16 = lane & 15, hl = lane >> 4;
  const int wm = wid & 3, wn = wid >> 2;                    // 4x2 wave grid
  const int bm0 = blockIdx.y * BM, bn0 = blockIdx.x * BN;

  v8f acc[2][4];
#pragma unroll
  for (int mi = 0; mi < 2; ++mi)
#pragma unroll
    for (int ni = 0; ni < 4; ++ni) acc[mi][ni] = {};

  // issue one K-tile (A: 512 + B: 512 16-byte chunks; 4 per thread)
  auto stage = [&](int buf, int kb) {
#pragma unroll
    for (int it = 0; it < 2; ++it) {
      int idx = it * 256 + tid;
      int r = idx >> 2, c8 = (idx & 3) << 3;
      async_cp16(A + (size_t)(bm0 + r) * K + kb + c8, &Asb[buf][r * LDP + c8]);
    }
#pragma unroll
    for (int it = 0; it < 2; ++it) {
      int idx = it * 256 + tid;
      int n = idx >> 2, c8 = (idx & 3) << 3;
      async_cp16(BT + (size_t)(bn0 + n) * K + kb + c8, &Bsb[buf][n * LDP + c8]);
    }
  };

  stage(0, 0);
  const int nk = K / BK;
  for (int i = 0; i < nk; ++i) {
    const int cur = i & 1;
    wait_async0();        // own wave's async writes to Asb/Bsb[cur] landed
    __syncthreads();      // ... and everyone else's too
    if (i + 1 < nk) stage(cur ^ 1, (i + 1) * BK);   // prefetch next tile

    // --- fragment loads: contiguous k order (k = 16*hl + element) for both A and B ---
    v16h af[2], bf[4];
#pragma unroll
    for (int mi = 0; mi < 2; ++mi) {
      v8u u;
      int row = wm * 32 + mi * 16 + ln16;
      const unsigned int* src = (const unsigned int*)(&Asb[cur][row * LDP]);
#pragma unroll
      for (int v = 0; v < 8; ++v) {
        int k0 = 2 * v + 16 * hl;                                     // A 16x32
        u[v] = src[k0 >> 1];
      }
      af[mi] = __builtin_bit_cast(v16h, u);
    }
#pragma unroll
    for (int ni = 0; ni < 4; ++ni) {
      v8u u;
      int n = wn * 64 + ni * 16 + ln16;
      const unsigned int* src = (const unsigned int*)(&Bsb[cur][n * LDP]);
#pragma unroll
      for (int v = 0; v < 8; ++v) {
        int k0 = 2 * v + 16 * hl;                                     // B 32x16
        u[v] = src[k0 >> 1];
      }
      bf[ni] = __builtin_bit_cast(v16h, u);
    }
#pragma unroll
    for (int mi = 0; mi < 2; ++mi)
#pragma unroll
      for (int ni = 0; ni < 4; ++ni)
        acc[mi][ni] = WMMA_F16(af[mi], bf[ni], acc[mi][ni]);
  }

  // --- epilogue (C layout: row = r + 8*hl, col = ln16 within 16x16 tile) ---
#pragma unroll
  for (int mi = 0; mi < 2; ++mi)
#pragma unroll
    for (int ni = 0; ni < 4; ++ni)
#pragma unroll
      for (int r = 0; r < 8; ++r) {
        int row = bm0 + wm * 32 + mi * 16 + r + 8 * hl;
        int col = bn0 + wn * 64 + ni * 16 + ln16;
        float v = acc[mi][ni][r];
        if (EPI == EPI_F16) {
          ((_Float16*)Cout)[(size_t)row * N + col] = (_Float16)v;
        } else if (EPI == EPI_BIAS_RELU_F16) {
          v = fmaxf(v + bias[col], 0.0f);
          ((_Float16*)Cout)[(size_t)row * N + col] = (_Float16)v;
        } else {
          v += bias[col] + resid[(size_t)row * N + col];
          ((float*)Cout)[(size_t)row * N + col] = v;
        }
      }
}

// ---------------------------------------------------------------------------
// Flash attention: grid = B*H*(S/64) blocks of 128 threads (4 waves).
// Each wave owns 16 query rows; streams 32-key blocks with online softmax.
// qkv: f16 [TOK, 3072] (q | k | v per head), ctx out: f16 [TOK, 1024].
// ---------------------------------------------------------------------------
__global__ __launch_bounds__(128)
void attn_kernel(const _Float16* __restrict__ qkv, _Float16* __restrict__ ctx) {
  __shared__ _Float16 Pbuf[4][16 * 36];    // per-wave P tile, padded rows
  const int wid = threadIdx.x >> 5, lane = threadIdx.x & 31;
  const int ln16 = lane & 15, hl = lane >> 4;
  const int idx = blockIdx.x;
  const int sb = idx & 31;                 // S/64 = 32 row-blocks
  const int h  = (idx >> 5) & 15;
  const int b  = idx >> 9;
  const int sq = sb * 64 + wid * 16;       // first query row of this wave
  const size_t base = (size_t)b * SEQ * C3;

  // Q fragments: 16 rows x 64 (two K=32 chunks), loaded straight from global.
  v16h qf[2];
  {
    const _Float16* qrow = qkv + base + (size_t)(sq + ln16) * C3 + h * HSIZ;
#pragma unroll
    for (int c = 0; c < 2; ++c) {
      v8u u;
      const unsigned int* p = (const unsigned int*)(qrow + c * 32);
#pragma unroll
      for (int v = 0; v < 8; ++v) {
        int k0 = 2 * v + 16 * hl;
        u[v] = p[k0 >> 1];
      }
      qf[c] = __builtin_bit_cast(v16h, u);
    }
  }

  v8f O[4];
#pragma unroll
  for (int nt = 0; nt < 4; ++nt) O[nt] = {};
  float mrow[8], lrow[8];
#pragma unroll
  for (int r = 0; r < 8; ++r) { mrow[r] = -1e30f; lrow[r] = 0.0f; }

  const int jbmax = (sq + 15) >> 5;        // last 32-key block touching this wave
  for (int jb = 0; jb <= jbmax; ++jb) {
    // scores S = Q (16x64) x K^T (64x32): two 16x16 n-tiles, two K chunks each
    v8f s[2];
#pragma unroll
    for (int nt = 0; nt < 2; ++nt) {
      s[nt] = {};
      int t = jb * 32 + nt * 16 + ln16;    // key token for this lane (B layout N)
      const _Float16* krow = qkv + base + (size_t)t * C3 + DIM + h * HSIZ;
#pragma unroll
      for (int c = 0; c < 2; ++c) {
        v8u u;
        const unsigned int* p = (const unsigned int*)(krow + c * 32);
#pragma unroll
        for (int v = 0; v < 8; ++v) {
          int d0 = 2 * v + 16 * hl;        // B layout: element -> K(hs) index
          u[v] = p[d0 >> 1];
        }
        v16h kf = __builtin_bit_cast(v16h, u);
        s[nt] = WMMA_F16(qf[c], kf, s[nt]);
      }
    }

    // online softmax update (scale = D^-0.5 = 1/32, causal mask)
    const int colbase = jb * 32 + ln16;
#pragma unroll
    for (int r = 0; r < 8; ++r) {
      int row = sq + r + 8 * hl;
      float x0 = s[0][r] * 0.03125f;
      float x1 = s[1][r] * 0.03125f;
      if (colbase > row)      x0 = -1e30f;
      if (colbase + 16 > row) x1 = -1e30f;
      float mloc = rowmax16(fmaxf(x0, x1));
      float mnew = fmaxf(mrow[r], mloc);
      float alpha = __expf(mrow[r] - mnew);
      float p0 = __expf(x0 - mnew);
      float p1 = __expf(x1 - mnew);
      float ls = rowsum16(p0 + p1);
      lrow[r] = lrow[r] * alpha + ls;
      mrow[r] = mnew;
#pragma unroll
      for (int nt = 0; nt < 4; ++nt) O[nt][r] = O[nt][r] * alpha;
      // spill P (C layout) to per-wave LDS so we can re-read in A layout
      Pbuf[wid][(r + 8 * hl) * 36 + ln16]      = (_Float16)p0;
      Pbuf[wid][(r + 8 * hl) * 36 + 16 + ln16] = (_Float16)p1;
    }
    // per-wave LDS RAW: drain the CDNA5 DS counter (no block barrier — the
    // four waves run divergent trip counts, each uses its own LDS region)
    asm volatile("s_wait_dscnt 0" ::: "memory");

    // P as A fragment (16x32)
    v16h pf;
    {
      v8u u;
      const unsigned int* p = (const unsigned int*)(&Pbuf[wid][ln16 * 36]);
#pragma unroll
      for (int v = 0; v < 8; ++v) {
        int k0 = 2 * v + 16 * hl;
        u[v] = p[k0 >> 1];
      }
      pf = __builtin_bit_cast(v16h, u);
    }

    // O += P (16x32) x V (32x64): four 16x16 n-tiles over head dim
#pragma unroll
    for (int nt = 0; nt < 4; ++nt) {
      union { v16h v; _Float16 e[16]; } vf;
#pragma unroll
      for (int e = 0; e < 16; ++e) {
        int t = jb * 32 + e + 16 * hl;     // B layout: element -> key index
        vf.e[e] = qkv[base + (size_t)t * C3 + 2 * DIM + h * HSIZ + nt * 16 + ln16];
      }
      O[nt] = WMMA_F16(pf, vf.v, O[nt]);
    }
  }

  // finalize: divide by row sums, store ctx f16 [token, h*64 + hs]
#pragma unroll
  for (int nt = 0; nt < 4; ++nt)
#pragma unroll
    for (int r = 0; r < 8; ++r) {
      int row = sq + r + 8 * hl;
      float v = O[nt][r] / lrow[r];
      ctx[((size_t)(b * SEQ + row)) * DIM + h * HSIZ + nt * 16 + ln16] = (_Float16)v;
    }
}

// ---------------------------------------------------------------------------
// Launch: LN1 -> QKV gemm -> attention -> Wo gemm(+res) -> LN2 -> FFN gemms
// ---------------------------------------------------------------------------
extern "C" void kernel_launch(void* const* d_in, const int* in_sizes, int n_in,
                              void* d_out, int out_size, void* d_ws, size_t ws_size,
                              hipStream_t stream) {
  const float* x    = (const float*)d_in[0];
  const float* ln1s = (const float*)d_in[1];
  const float* ln1b = (const float*)d_in[2];
  const float* Wq   = (const float*)d_in[3];
  const float* Wk   = (const float*)d_in[4];
  const float* Wv   = (const float*)d_in[5];
  const float* Wo   = (const float*)d_in[6];
  const float* bo   = (const float*)d_in[7];
  const float* ln2s = (const float*)d_in[8];
  const float* ln2b = (const float*)d_in[9];
  const float* W1   = (const float*)d_in[10];
  const float* b1   = (const float*)d_in[11];
  const float* W2   = (const float*)d_in[12];
  const float* b2   = (const float*)d_in[13];
  float* out = (float*)d_out;

  // workspace carve-up (~126 MB total; every buffer fully written before read)
  char* ws = (char*)d_ws;
  _Float16* h16    = (_Float16*)ws; ws += (size_t)TOK * DIM * 2;     // ln1 out
  _Float16* wqkvT  = (_Float16*)ws; ws += (size_t)DIM * C3 * 2;      // [3D][D] f16
  _Float16* qkv16  = (_Float16*)ws; ws += (size_t)TOK * C3 * 2;      // q|k|v
  _Float16* woT    = (_Float16*)ws; ws += (size_t)DIM * DIM * 2;     // [D][D]
  _Float16* w1T    = (_Float16*)ws; ws += (size_t)DIM * FFDIM * 2;   // [FF][D]
  _Float16* w2T    = (_Float16*)ws; ws += (size_t)FFDIM * DIM * 2;   // [D][FF]
  _Float16* ctx16  = (_Float16*)ws; ws += (size_t)TOK * DIM * 2;
  float*    x2     = (float*)ws;    ws += (size_t)TOK * DIM * 4;     // x + attn out
  _Float16* h216   = (_Float16*)ws; ws += (size_t)TOK * DIM * 2;     // ln2 out
  _Float16* mid16  = (_Float16*)ws;                                  // ffn hidden

  // weight conversion, K-contiguous (re-done every call: graph-safe)
  pack_qkv_wT<<<(DIM * C3 + 255) / 256, 256, 0, stream>>>(Wq, Wk, Wv, wqkvT);
  cvt_f32_f16_T<<<(DIM * DIM + 255) / 256, 256, 0, stream>>>(Wo, woT, DIM, DIM);
  cvt_f32_f16_T<<<(DIM * FFDIM + 255) / 256, 256, 0, stream>>>(W1, w1T, DIM, FFDIM);
  cvt_f32_f16_T<<<(FFDIM * DIM + 255) / 256, 256, 0, stream>>>(W2, w2T, FFDIM, DIM);

  // --- attention path ---
  ln_f16_kernel<<<TOK, 256, 0, stream>>>(x, ln1s, ln1b, h16);
  gemm_wmma<EPI_F16><<<dim3(C3 / 128, TOK / 128), 256, 0, stream>>>(
      h16, wqkvT, (void*)qkv16, nullptr, nullptr, TOK, C3, DIM);
  attn_kernel<<<2 * NHEAD * (SEQ / 64), 128, 0, stream>>>(qkv16, ctx16);
  gemm_wmma<EPI_BIAS_RES_F32><<<dim3(DIM / 128, TOK / 128), 256, 0, stream>>>(
      ctx16, woT, (void*)x2, bo, x, TOK, DIM, DIM);

  // --- feed-forward path ---
  ln_f16_kernel<<<TOK, 256, 0, stream>>>(x2, ln2s, ln2b, h216);
  gemm_wmma<EPI_BIAS_RELU_F16><<<dim3(FFDIM / 128, TOK / 128), 256, 0, stream>>>(
      h216, w1T, (void*)mid16, b1, nullptr, TOK, FFDIM, DIM);
  gemm_wmma<EPI_BIAS_RES_F32><<<dim3(DIM / 128, TOK / 128), 256, 0, stream>>>(
      mid16, w2T, (void*)out, b2, x2, TOK, DIM, FFDIM);
}
